// MultiHeadSelfAttention_36679020708610
// MI455X (gfx1250) — compile-verified, matrix operand order checked by reading, not run on hardware
//
#include <hip/hip_runtime.h>
#include <hip/hip_bf16.h>

// ---------- types ----------
typedef __bf16 v8bf  __attribute__((ext_vector_type(8)));
typedef __bf16 v16bf __attribute__((ext_vector_type(16)));
typedef float  v8f   __attribute__((ext_vector_type(8)));

// ---------- helpers ----------
__device__ __forceinline__ unsigned short bf16bits(float f) {
    unsigned u = __float_as_uint(f);
    u += 0x7FFFu + ((u >> 16) & 1u);      // round to nearest even
    return (unsigned short)(u >> 16);
}

__device__ __forceinline__ __bf16 to_bf16(float f) {
    unsigned short s = bf16bits(f);
    return __builtin_bit_cast(__bf16, s);
}

// Per-lane WMMA A/B operand slice: two contiguous 8-elem bf16 chunks
// (16B each -> one s_clause of two global_load_b128).
__device__ __forceinline__ v16bf ld_ab(const __bf16* p0, const __bf16* p1) {
    v8bf lo = *reinterpret_cast<const v8bf*>(p0);
    v8bf hi = *reinterpret_cast<const v8bf*>(p1);
    v16bf r;
#pragma unroll
    for (int i = 0; i < 8; ++i) { r[i] = lo[i]; r[i + 8] = hi[i]; }
    return r;
}

__device__ __forceinline__ v8f wmma_bf16(v16bf a, v16bf b, v8f c) {
    return __builtin_amdgcn_wmma_f32_16x16x32_bf16(
        /*neg_a=*/false, a, /*neg_b=*/false, b,
        /*c_mod=*/(short)0, c, /*reuse_a=*/false, /*reuse_b=*/false);
}

#define E_DIM 1024
#define T_DIM 2048
#define H_DIM 16
#define D_DIM 64

// ---------- 0) fp32 -> bf16 conversion ----------
__global__ void mhsa_cvt_bf16(const float* __restrict__ s,
                              unsigned short* __restrict__ d, int n) {
    int i = blockIdx.x * blockDim.x + threadIdx.x;
    int stride = gridDim.x * blockDim.x;
    for (; i < n; i += stride) d[i] = bf16bits(s[i]);
}

// Load one k-step of operand tiles: 2 A row-tiles (32 rows) + 4 B col-tiles.
__device__ __forceinline__ void load_step(const __bf16* arow0,
                                          const __bf16* arow1,
                                          const __bf16* wr0, int kb, int c,
                                          v16bf& a0, v16bf& a1, v16bf* b) {
    a0 = ld_ab(arow0 + kb + c, arow0 + kb + c + 16);
    a1 = ld_ab(arow1 + kb + c, arow1 + kb + c + 16);
#pragma unroll
    for (int nt = 0; nt < 4; ++nt) {
        const __bf16* wrow = wr0 + (size_t)(nt * 16) * E_DIM;
        b[nt] = ld_ab(wrow + kb + c, wrow + kb + c + 16);
    }
}

// ---------- 1) QKV GEMM: [4096,1024] x [3072,1024]^T + b -> scatter Q/K/Vt ----
// grid (N/128, M/128), 256 threads (8 waves); wave -> 32x64 C tile,
// software-pipelined (double-buffered operand registers).
__global__ __launch_bounds__(256) void mhsa_qkv_gemm(
    const unsigned short* __restrict__ xb, const unsigned short* __restrict__ wb,
    const float* __restrict__ bias,
    unsigned short* __restrict__ Qb, unsigned short* __restrict__ Kb,
    unsigned short* __restrict__ Vt) {
    const __bf16* X = reinterpret_cast<const __bf16*>(xb);
    const __bf16* W = reinterpret_cast<const __bf16*>(wb);
    const int wave = threadIdx.x >> 5, lane = threadIdx.x & 31;
    const int rt = lane & 15, grp = lane >> 4, c = grp * 8;
    const int wm = wave & 3, wn = wave >> 2;
    const int m0 = blockIdx.y * 128 + wm * 32;
    const int n0 = blockIdx.x * 128 + wn * 64;

    v8f acc[2][4] = {};
    const __bf16* arow0 = X + (size_t)(m0 + rt) * E_DIM;
    const __bf16* arow1 = arow0 + (size_t)16 * E_DIM;
    const __bf16* wr0 = W + (size_t)(n0 + rt) * E_DIM;

    v16bf A0[2], A1[2], Bt[2][4];
    load_step(arow0, arow1, wr0, 0, c, A0[0], A1[0], Bt[0]);
#pragma unroll 2
    for (int it = 0; it < E_DIM / 32; ++it) {
        const int cur = it & 1, nxt = cur ^ 1;
        if (it + 1 < E_DIM / 32)   // prefetch next k-step while WMMAs run
            load_step(arow0, arow1, wr0, (it + 1) * 32, c,
                      A0[nxt], A1[nxt], Bt[nxt]);
#pragma unroll
        for (int nt = 0; nt < 4; ++nt) {
            acc[0][nt] = wmma_bf16(A0[cur], Bt[cur][nt], acc[0][nt]);
            acc[1][nt] = wmma_bf16(A1[cur], Bt[cur][nt], acc[1][nt]);
        }
    }

#pragma unroll
    for (int nt = 0; nt < 4; ++nt) {
        const int n = n0 + nt * 16 + rt;
        const float bv = bias[n];
        const int sect = n >> 10;            // 0=q 1=k 2=v
        const int e = n & 1023, h = e >> 6, d = e & 63;
#pragma unroll
        for (int mt = 0; mt < 2; ++mt) {
#pragma unroll
            for (int r = 0; r < 8; ++r) {
                const int m = m0 + mt * 16 + r + grp * 8;
                const int bi = m >> 11, t = m & 2047;
                const float val = acc[mt][nt][r] + bv;
                const size_t head = (size_t)(bi * H_DIM + h);
                if (sect == 0)    // fold 1/sqrt(D) into Q
                    Qb[(head * T_DIM + t) * D_DIM + d] = bf16bits(val * 0.125f);
                else if (sect == 1)
                    Kb[(head * T_DIM + t) * D_DIM + d] = bf16bits(val);
                else              // V stored transposed [BH][D][T]
                    Vt[(head * D_DIM + d) * T_DIM + t] = bf16bits(val);
            }
        }
    }
}

// ---------- 2) Flash attention: grid (BH=32, T/128=16), wave -> 16 query rows --
__global__ __launch_bounds__(256) void mhsa_attention(
    const unsigned short* __restrict__ Qb_, const unsigned short* __restrict__ Kb_,
    const unsigned short* __restrict__ Vt_, unsigned short* __restrict__ Ob_) {
    __shared__ float ptile[8][16][33];     // per-wave C->A transpose scratch
    const __bf16* Q = reinterpret_cast<const __bf16*>(Qb_);
    const __bf16* K = reinterpret_cast<const __bf16*>(Kb_);
    const __bf16* V = reinterpret_cast<const __bf16*>(Vt_);

    const int wave = threadIdx.x >> 5, lane = threadIdx.x & 31;
    const int rt = lane & 15, grp = lane >> 4, c = grp * 8;
    const int bh = blockIdx.x;
    const int q0 = blockIdx.y * 128 + wave * 16;

    const __bf16* qrow = Q + ((size_t)bh * T_DIM + q0 + rt) * D_DIM;
    const v16bf qa0 = ld_ab(qrow + c, qrow + c + 16);            // D 0..31
    const v16bf qa1 = ld_ab(qrow + 32 + c, qrow + 32 + c + 16);  // D 32..63

    v8f acc[4] = {};
    float mrow[8], lrow[8];
#pragma unroll
    for (int r = 0; r < 8; ++r) { mrow[r] = -3.0e38f; lrow[r] = 0.0f; }

    for (int kb = 0; kb < T_DIM; kb += 32) {
        // ---- S = Q @ K^T for two 16-wide key tiles ----
        const __bf16* kr0 = K + ((size_t)bh * T_DIM + kb + rt) * D_DIM;
        const __bf16* kr1 = kr0 + (size_t)16 * D_DIM;
        if (kb + 32 < T_DIM)   // touch next key block (global_prefetch)
            __builtin_prefetch(kr0 + (size_t)32 * D_DIM, 0, 0);
        v8f z = {};
        v8f s0 = wmma_bf16(qa0, ld_ab(kr0 + c, kr0 + c + 16), z);
        s0 = wmma_bf16(qa1, ld_ab(kr0 + 32 + c, kr0 + 32 + c + 16), s0);
        v8f s1 = wmma_bf16(qa0, ld_ab(kr1 + c, kr1 + c + 16), z);
        s1 = wmma_bf16(qa1, ld_ab(kr1 + 32 + c, kr1 + 32 + c + 16), s1);

        // ---- issue V tile loads now; they fly during the softmax VALU burst --
        v16bf vb[4];
#pragma unroll
        for (int nt = 0; nt < 4; ++nt) {
            const __bf16* vrow =
                V + ((size_t)bh * D_DIM + nt * 16 + rt) * T_DIM + kb;
            vb[nt] = ld_ab(vrow + c, vrow + c + 16);
        }

        // ---- streaming softmax update (row = r + grp*8, col = lane&15) ----
#pragma unroll
        for (int r = 0; r < 8; ++r) {
            float v0 = s0[r], v1 = s1[r];
            float mx = fmaxf(v0, v1);
#pragma unroll
            for (int msk = 1; msk <= 8; msk <<= 1)
                mx = fmaxf(mx, __shfl_xor(mx, msk, 32));
            const float mnew = fmaxf(mrow[r], mx);
            const float corr = __expf(mrow[r] - mnew);
            const float p0 = __expf(v0 - mnew);
            const float p1 = __expf(v1 - mnew);
            float rs = p0 + p1;
#pragma unroll
            for (int msk = 1; msk <= 8; msk <<= 1)
                rs += __shfl_xor(rs, msk, 32);
            lrow[r] = lrow[r] * corr + rs;
            mrow[r] = mnew;
#pragma unroll
            for (int nt = 0; nt < 4; ++nt) acc[nt][r] *= corr;
            const int row = r + grp * 8;
            ptile[wave][row][rt] = p0;
            ptile[wave][row][16 + rt] = p1;
        }
        __syncthreads();

        // ---- P (C layout) -> A layout through LDS ----
        v16bf pa;
#pragma unroll
        for (int j = 0; j < 16; ++j) {
            const int col = (j < 8 ? j : j + 8) + c;
            pa[j] = to_bf16(ptile[wave][rt][col]);
        }

        // ---- O += P @ V ----
#pragma unroll
        for (int nt = 0; nt < 4; ++nt)
            acc[nt] = wmma_bf16(pa, vb[nt], acc[nt]);
        __syncthreads();
    }

    // ---- normalize and merge heads: out[b][t][h*64+d] ----
    const int b = bh >> 4, h = bh & 15;
#pragma unroll
    for (int nt = 0; nt < 4; ++nt) {
        const int e = h * D_DIM + nt * 16 + rt;
#pragma unroll
        for (int r = 0; r < 8; ++r) {
            const int t = q0 + r + grp * 8;
            const float val = acc[nt][r] / lrow[r];
            Ob_[((size_t)(b * T_DIM + t)) * E_DIM + e] = bf16bits(val);
        }
    }
}

// ---------- 3) Output GEMM: [4096,1024] x [1024,1024]^T + b -> fp32 out -------
__global__ __launch_bounds__(256) void mhsa_out_gemm(
    const unsigned short* __restrict__ ab, const unsigned short* __restrict__ wb,
    const float* __restrict__ bias, float* __restrict__ out) {
    const __bf16* A = reinterpret_cast<const __bf16*>(ab);
    const __bf16* W = reinterpret_cast<const __bf16*>(wb);
    const int wave = threadIdx.x >> 5, lane = threadIdx.x & 31;
    const int rt = lane & 15, grp = lane >> 4, c = grp * 8;
    const int wm = wave & 3, wn = wave >> 2;
    const int m0 = blockIdx.y * 128 + wm * 32;
    const int n0 = blockIdx.x * 128 + wn * 64;

    v8f acc[2][4] = {};
    const __bf16* arow0 = A + (size_t)(m0 + rt) * E_DIM;
    const __bf16* arow1 = arow0 + (size_t)16 * E_DIM;
    const __bf16* wr0 = W + (size_t)(n0 + rt) * E_DIM;

    v16bf A0[2], A1[2], Bt[2][4];
    load_step(arow0, arow1, wr0, 0, c, A0[0], A1[0], Bt[0]);
#pragma unroll 2
    for (int it = 0; it < E_DIM / 32; ++it) {
        const int cur = it & 1, nxt = cur ^ 1;
        if (it + 1 < E_DIM / 32)
            load_step(arow0, arow1, wr0, (it + 1) * 32, c,
                      A0[nxt], A1[nxt], Bt[nxt]);
#pragma unroll
        for (int nt = 0; nt < 4; ++nt) {
            acc[0][nt] = wmma_bf16(A0[cur], Bt[cur][nt], acc[0][nt]);
            acc[1][nt] = wmma_bf16(A1[cur], Bt[cur][nt], acc[1][nt]);
        }
    }

#pragma unroll
    for (int nt = 0; nt < 4; ++nt) {
        const int n = n0 + nt * 16 + rt;
        const float bv = bias[n];
#pragma unroll
        for (int mt = 0; mt < 2; ++mt) {
#pragma unroll
            for (int r = 0; r < 8; ++r) {
                const int m = m0 + mt * 16 + r + grp * 8;
                out[(size_t)m * E_DIM + n] = acc[mt][nt][r] + bv;
            }
        }
    }
}

// ---------- launcher ----------
extern "C" void kernel_launch(void* const* d_in, const int* in_sizes, int n_in,
                              void* d_out, int out_size, void* d_ws, size_t ws_size,
                              hipStream_t stream) {
    const float* x     = (const float*)d_in[0];
    const float* qkv_w = (const float*)d_in[1];
    const float* qkv_b = (const float*)d_in[2];
    const float* out_w = (const float*)d_in[3];
    const float* out_b = (const float*)d_in[4];
    float* out = (float*)d_out;

    const int M  = 2 * T_DIM;           // 4096 rows (B*T)
    const int BH = 2 * H_DIM;           // 32 head-batches

    // carve bf16 workspace (ushort bit-pattern buffers), 256B aligned
    char* ws = (char*)d_ws;
    size_t off = 0;
    auto carve = [&](size_t elems) {
        unsigned short* p = (unsigned short*)(ws + off);
        off = (off + elems * sizeof(unsigned short) + 255) & ~(size_t)255;
        return p;
    };
    unsigned short* xb    = carve((size_t)M * E_DIM);            // x bf16
    unsigned short* wqkvb = carve((size_t)3 * E_DIM * E_DIM);    // qkv_w bf16
    unsigned short* wob   = carve((size_t)E_DIM * E_DIM);        // out_w bf16
    unsigned short* Qb    = carve((size_t)BH * T_DIM * D_DIM);   // [BH][T][D], pre-scaled
    unsigned short* Kb    = carve((size_t)BH * T_DIM * D_DIM);   // [BH][T][D]
    unsigned short* Vt    = carve((size_t)BH * D_DIM * T_DIM);   // [BH][D][T]
    unsigned short* Ob    = carve((size_t)M * E_DIM);            // merged heads bf16

    // 0) conversions
    mhsa_cvt_bf16<<<4096, 256, 0, stream>>>(x, xb, M * E_DIM);
    mhsa_cvt_bf16<<<4096, 256, 0, stream>>>(qkv_w, wqkvb, 3 * E_DIM * E_DIM);
    mhsa_cvt_bf16<<<4096, 256, 0, stream>>>(out_w, wob, E_DIM * E_DIM);

    // 1) QKV projection -> Q / K / V^T (bf16, bias fused, Q pre-scaled)
    mhsa_qkv_gemm<<<dim3(3 * E_DIM / 128, M / 128), 256, 0, stream>>>(
        xb, wqkvb, qkv_b, Qb, Kb, Vt);

    // 2) streaming-softmax attention, heads merged on store
    mhsa_attention<<<dim3(BH, T_DIM / 128), 256, 0, stream>>>(Qb, Kb, Vt, Ob);

    // 3) output projection (fp32 result + bias)
    mhsa_out_gemm<<<dim3(E_DIM / 128, M / 128), 256, 0, stream>>>(
        Ob, wob, out_b, out);
}
